// Decoder_12317966205227
// MI455X (gfx1250) — matrix operand order repaired and checked by reading, not compiled, not run on hardware
//
#include <hip/hip_runtime.h>
#include <hip/hip_bf16.h>

typedef __attribute__((ext_vector_type(16))) _Float16 v16h;
typedef __attribute__((ext_vector_type(8)))  _Float16 v8h;
typedef __attribute__((ext_vector_type(8)))  float    v8f;

#define HH     512
#define EE     512
#define BB     16
#define TDD    64
#define TEE    256
#define VVOC   50257
#define NPADFC 50688   /* 99 * 512 */
#define G4H    2048
#define XK0    1536    /* E + H + H  (emb, context, h0) */
#define XK1    1024    /* H + H      (h0_new, h1_prev)  */

__device__ __forceinline__ float sigm(float x) { return 1.0f / (1.0f + __expf(-x)); }

// ---------------------------------------------------------------------------
// Generic f16 WMMA GEMM:  C(M,Nstore) = A(M,K) * Wt(Npad,K)^T + bias
// grid = (Npad/512, M/(16*MT)), block = 256 (8 waves).
// Each wave: MT M-tiles x 4 N-tiles (register blocking; A frags reused
// across N, B frags reused across M -> L2 traffic / MT).
// K is a compile-time constant so all k-loop addressing folds into
// global_load immediate offsets. __launch_bounds__(256,1) raises the VGPR
// budget so the MT=4 accumulator set (128 VGPRs) does not spill.
// A/B fragments use contiguous k (lane half h: K 16h..16h+15); C/D per ISA (wave32).
// ---------------------------------------------------------------------------
template<int MT, int K, bool NT>
__global__ __launch_bounds__(256, 1) void wmma_gemm(
    const _Float16* __restrict__ A,
    const _Float16* __restrict__ Wt,
    const float* __restrict__ bias,
    float* __restrict__ C,
    int ldc, int Nstore)
{
    const int lane = threadIdx.x & 31;
    const int wave = threadIdx.x >> 5;
    const int m0   = blockIdx.y * (16 * MT);
    const int nt0  = blockIdx.x * 32 + wave * 4;   // first of 4 N-tiles

    v8f acc[MT][4];
    #pragma unroll
    for (int mt = 0; mt < MT; ++mt)
        #pragma unroll
        for (int i = 0; i < 4; ++i)
            #pragma unroll
            for (int r = 0; r < 8; ++r) acc[mt][i][r] = 0.0f;

    const int half16 = (lane < 16);
    // Per-wave base pointers; all k-loop offsets are compile-time constants.
    const _Float16* aBase = A + ((size_t)(m0 + (lane & 15))) * K + (half16 ? 0 : 16);
    const _Float16* bBase = Wt + ((size_t)(nt0 * 16 + (lane & 15))) * K + (half16 ? 0 : 16);

    for (int k0 = 0; k0 < K; k0 += 32) {
        // A fragment 16x32 f16: lane<16 -> K {0..15}; lane>=16 -> {16..31}
        v16h a[MT];
        #pragma unroll
        for (int mt = 0; mt < MT; ++mt) {
            const v8h lo = *(const v8h*)(aBase + (size_t)mt * 16 * K + k0);
            const v8h hi = *(const v8h*)(aBase + (size_t)mt * 16 * K + k0 + 8);
            #pragma unroll
            for (int i = 0; i < 8; ++i) { a[mt][i] = lo[i]; a[mt][i + 8] = hi[i]; }
        }
        #pragma unroll
        for (int i = 0; i < 4; ++i) {
            // B fragment 32x16: lane n holds 16 consecutive K (0..15 or 16..31)
            const v16h b = *(const v16h*)(bBase + (size_t)i * 16 * K + k0);
            #pragma unroll
            for (int mt = 0; mt < MT; ++mt)
                acc[mt][i] = __builtin_amdgcn_wmma_f32_16x16x32_f16(
                    false, a[mt], false, b, (short)0, acc[mt][i], false, false);
        }
    }

    // C/D layout: VGPR r holds M = tileM + r (+8 for lanes 16..31), N = lane%16
    #pragma unroll
    for (int i = 0; i < 4; ++i) {
        const int col = (nt0 + i) * 16 + (lane & 15);
        if (col < Nstore) {
            const float bv = bias ? bias[col] : 0.0f;
            #pragma unroll
            for (int mt = 0; mt < MT; ++mt) {
                const int rbase = m0 + mt * 16 + (half16 ? 0 : 8);
                #pragma unroll
                for (int r = 0; r < 8; ++r) {
                    float* dst = C + (size_t)(rbase + r) * ldc + col;
                    const float val = acc[mt][i][r] + bv;
                    if (NT) __builtin_nontemporal_store(val, dst);
                    else    *dst = val;
                }
            }
        }
    }
}

// ---------------------------------------------------------------------------
// Fused (previous-step layer-1 cell) + attention + input packing.
// grid=16 (one block per batch), block=256.
// Block b first finishes the layer-1 LSTM cell for its own batch (h1 only
// feeds batch-local attention -> no cross-block dependency), keeping h1 in
// LDS, then computes attention and writes xcat0 = [emb_t | ctx | h0] (f16).
// ---------------------------------------------------------------------------
__global__ void attn_fused_kernel(float* __restrict__ h1,
                                  float* __restrict__ c1,
                                  const float* __restrict__ g1,
                                  _Float16* __restrict__ hs16,
                                  const float* __restrict__ h0,
                                  const float* __restrict__ w2t,  // (K,N) = W2^T
                                  const float* __restrict__ W2b,
                                  const float* __restrict__ encT, // (b,k,e)
                                  const float* __restrict__ encout,
                                  const int*   __restrict__ srcmask,
                                  const float* __restrict__ vw,
                                  const float* __restrict__ vb,
                                  const int*   __restrict__ ids,
                                  const float* __restrict__ emb,
                                  _Float16* __restrict__ xcat0,
                                  int t)
{
    __shared__ float h1s[HH];
    __shared__ float q[HH];
    __shared__ float sc[TEE];
    __shared__ float red[TEE];
    const int b   = blockIdx.x;
    const int tid = threadIdx.x;

    // ---- phase 0: layer-1 cell update for step t-1 (or load h1 if t==0) ----
    if (t > 0) {
        const float* gb = g1 + (size_t)b * G4H;
        for (int j = tid; j < HH; j += 256) {
            const int idx = b * HH + j;
            const float ii = sigm(gb[j]);
            const float ff = sigm(gb[512 + j]);
            const float gg = tanhf(gb[1024 + j]);
            const float oo = sigm(gb[1536 + j]);
            const float cn = ff * c1[idx] + ii * gg;
            const float hn = oo * tanhf(cn);
            c1[idx] = cn; h1[idx] = hn; h1s[j] = hn;
            hs16[((size_t)(b * TDD + (t - 1))) * HH + j] = (_Float16)hn;
        }
    } else {
        for (int j = tid; j < HH; j += 256) h1s[j] = h1[b * HH + j];
    }
    __syncthreads();

    // ---- q = h1 @ W2^T + b2  (coalesced via transposed W2) ----
    for (int j = tid; j < HH; j += 256) {
        float s = W2b[j];
        for (int k = 0; k < HH; ++k) s += h1s[k] * w2t[(size_t)k * HH + j];
        q[j] = s;
    }
    __syncthreads();

    // ---- score_e = v . tanh(enc_proj[b,:,e] + q) + vb  (masked) ----
    {
        const float* ep = encT + (size_t)b * HH * TEE;
        float s = vb[0];
        for (int k = 0; k < HH; ++k) s += tanhf(ep[(size_t)k * TEE + tid] + q[k]) * vw[k];
        if (srcmask[b * TEE + tid] == 0) s = -1e9f;
        sc[tid] = s; red[tid] = s;
    }
    __syncthreads();
    for (int off = 128; off > 0; off >>= 1) {
        if (tid < off) red[tid] = fmaxf(red[tid], red[tid + off]);
        __syncthreads();
    }
    const float mx = red[0];
    __syncthreads();
    const float ex = __expf(sc[tid] - mx);
    sc[tid] = ex; red[tid] = ex;
    __syncthreads();
    for (int off = 128; off > 0; off >>= 1) {
        if (tid < off) red[tid] += red[tid + off];
        __syncthreads();
    }
    const float inv = 1.0f / red[0];
    __syncthreads();

    // ---- context + pack [emb_t | ctx | h0] ----
    _Float16* xb = xcat0 + (size_t)b * XK0;
    const int id = ids[b * TDD + t];
    const float* er = emb + (size_t)id * EE;
    for (int h = tid; h < HH; h += 256) {
        float ctx = 0.0f;
        const float* eb = encout + ((size_t)b * TEE) * HH + h;
        for (int e = 0; e < TEE; ++e) ctx += sc[e] * eb[(size_t)e * HH];
        xb[h]           = (_Float16)er[h];
        xb[EE + h]      = (_Float16)(ctx * inv);
        xb[EE + HH + h] = (_Float16)h0[b * HH + h];
    }
}

// ---------------------------------------------------------------------------
// LSTM cell pointwise update (PyTorch gate order i,f,g,o). 8192 threads.
// layer 0: packs xcat1 = [h0_new | h1_prev]; layer 1: writes hs_f16 row b*T+t.
// ---------------------------------------------------------------------------
__global__ void cell_kernel(const float* __restrict__ g,
                            float* __restrict__ hbuf,
                            float* __restrict__ cbuf,
                            const float* __restrict__ hother,
                            _Float16* __restrict__ xnext,
                            int layer, int t)
{
    const int idx = blockIdx.x * blockDim.x + threadIdx.x;  // 0..8191
    const int b = idx >> 9;
    const int j = idx & 511;
    const float* gb = g + (size_t)b * G4H;
    const float ii = sigm(gb[j]);
    const float ff = sigm(gb[512 + j]);
    const float gg = tanhf(gb[1024 + j]);
    const float oo = sigm(gb[1536 + j]);
    const float cn = ff * cbuf[idx] + ii * gg;
    const float hn = oo * tanhf(cn);
    cbuf[idx] = cn;
    hbuf[idx] = hn;
    if (layer == 0) {
        xnext[(size_t)b * XK1 + j]        = (_Float16)hn;
        xnext[(size_t)b * XK1 + 512 + j]  = (_Float16)hother[idx];
    } else {
        xnext[((size_t)(b * TDD + t)) * HH + j] = (_Float16)hn;
    }
}

// --------------------------- prep kernels ----------------------------------
__global__ void k_f32_to_f16(const float* __restrict__ in, _Float16* __restrict__ out, int n) {
    int i = blockIdx.x * blockDim.x + threadIdx.x;
    if (i < n) out[i] = (_Float16)in[i];
}

__global__ void k_fcw_conv(const float* __restrict__ fcw, _Float16* __restrict__ out) {
    size_t i = (size_t)blockIdx.x * blockDim.x + threadIdx.x;   // over NPADFC*512
    if (i >= (size_t)NPADFC * 512) return;
    int row = (int)(i >> 9);
    int col = (int)(i & 511);
    out[i] = (row < VVOC) ? (_Float16)fcw[(size_t)row * 512 + col] : (_Float16)0.0f;
}

__global__ void k_wcat(const float* __restrict__ Wi, const float* __restrict__ Wh,
                       _Float16* __restrict__ out, int kin) {
    const int ktot = kin + 512;
    size_t i = (size_t)blockIdx.x * blockDim.x + threadIdx.x;
    if (i >= (size_t)G4H * ktot) return;
    int j = (int)(i / ktot);
    int c = (int)(i % ktot);
    float v = (c < kin) ? Wi[(size_t)j * kin + c] : Wh[(size_t)j * 512 + (c - kin)];
    out[i] = (_Float16)v;
}

__global__ void k_bias(const float* bi0, const float* bh0,
                       const float* bi1, const float* bh1,
                       float* o0, float* o1) {
    int i = blockIdx.x * blockDim.x + threadIdx.x;
    if (i < G4H) { o0[i] = bi0[i] + bh0[i]; o1[i] = bi1[i] + bh1[i]; }
}

__global__ void k_init(const float* __restrict__ ih, const float* __restrict__ ic,
                       float* h0, float* c0, float* h1, float* c1) {
    int i = blockIdx.x * blockDim.x + threadIdx.x;
    if (i < BB * HH) {
        h0[i] = ih[i]; h1[i] = ih[BB * HH + i];
        c0[i] = ic[i]; c1[i] = ic[BB * HH + i];
    }
}

// transpose 512x512: out[k*512+j] = in[j*512+k]
__global__ void k_t512(const float* __restrict__ in, float* __restrict__ out) {
    int i = blockIdx.x * blockDim.x + threadIdx.x;
    if (i < HH * HH) {
        int k = i >> 9, j = i & 511;
        out[i] = in[(size_t)j * HH + k];
    }
}

// enc_proj (b,e,k) -> encT (b,k,e)
__global__ void k_encT(const float* __restrict__ in, float* __restrict__ out) {
    int i = blockIdx.x * blockDim.x + threadIdx.x;   // over 16*512*256
    if (i < BB * HH * TEE) {
        int e = i & (TEE - 1);
        int k = (i >> 8) & (HH - 1);
        int b = i >> 17;
        out[i] = in[((size_t)(b * TEE + e)) * HH + k];
    }
}

// ---------------------------------------------------------------------------
static inline size_t alignup(size_t x) { return (x + 255) & ~(size_t)255; }

extern "C" void kernel_launch(void* const* d_in, const int* in_sizes, int n_in,
                              void* d_out, int out_size, void* d_ws, size_t ws_size,
                              hipStream_t stream)
{
    (void)in_sizes; (void)n_in; (void)out_size; (void)ws_size;

    const int*   ids    = (const int*)  d_in[0];
    const float* init_h = (const float*)d_in[1];
    const float* init_c = (const float*)d_in[2];
    const float* encout = (const float*)d_in[3];
    const int*   smask  = (const int*)  d_in[4];
    const float* emb    = (const float*)d_in[5];
    const float* W1w    = (const float*)d_in[6];
    const float* W1b    = (const float*)d_in[7];
    const float* W2w    = (const float*)d_in[8];
    const float* W2b    = (const float*)d_in[9];
    const float* vw     = (const float*)d_in[10];
    const float* vb     = (const float*)d_in[11];
    const float* Wi0    = (const float*)d_in[12];
    const float* Wh0    = (const float*)d_in[13];
    const float* bi0    = (const float*)d_in[14];
    const float* bh0    = (const float*)d_in[15];
    const float* Wi1    = (const float*)d_in[16];
    const float* Wh1    = (const float*)d_in[17];
    const float* bi1    = (const float*)d_in[18];
    const float* bh1    = (const float*)d_in[19];
    const float* fcw    = (const float*)d_in[20];
    const float* fcb    = (const float*)d_in[21];
    float* out = (float*)d_out;

    // ---- workspace carve-up ----
    char* p = (char*)d_ws;
    size_t off = 0;
    auto take = [&](size_t bytes) { void* r = p + off; off = alignup(off + bytes); return r; };

    _Float16* fcw16   = (_Float16*)take((size_t)NPADFC * 512 * 2);
    _Float16* enc16   = (_Float16*)take((size_t)BB * TEE * HH * 2);
    _Float16* w116    = (_Float16*)take((size_t)HH * HH * 2);
    float*    encproj = (float*)   take((size_t)BB * TEE * HH * 4);
    float*    encT    = (float*)   take((size_t)BB * TEE * HH * 4);
    float*    w2t     = (float*)   take((size_t)HH * HH * 4);
    _Float16* wcat0   = (_Float16*)take((size_t)G4H * XK0 * 2);
    _Float16* wcat1   = (_Float16*)take((size_t)G4H * XK1 * 2);
    float*    bias0   = (float*)   take((size_t)G4H * 4);
    float*    bias1   = (float*)   take((size_t)G4H * 4);
    _Float16* xcat0   = (_Float16*)take((size_t)BB * XK0 * 2);
    _Float16* xcat1   = (_Float16*)take((size_t)BB * XK1 * 2);
    float*    g0      = (float*)   take((size_t)BB * G4H * 4);
    float*    g1      = (float*)   take((size_t)BB * G4H * 4);
    float*    h0      = (float*)   take((size_t)BB * HH * 4);
    float*    c0      = (float*)   take((size_t)BB * HH * 4);
    float*    h1      = (float*)   take((size_t)BB * HH * 4);
    float*    c1      = (float*)   take((size_t)BB * HH * 4);
    _Float16* hs16    = (_Float16*)take((size_t)BB * TDD * HH * 2);

    // ---- one-time prep (re-done every call: deterministic) ----
    {
        int n = BB * TEE * HH;
        k_f32_to_f16<<<(n + 255) / 256, 256, 0, stream>>>(encout, enc16, n);
        n = HH * HH;
        k_f32_to_f16<<<(n + 255) / 256, 256, 0, stream>>>(W1w, w116, n);
        k_t512<<<(HH * HH + 255) / 256, 256, 0, stream>>>(W2w, w2t);
        size_t nf = (size_t)NPADFC * 512;
        k_fcw_conv<<<(unsigned)((nf + 255) / 256), 256, 0, stream>>>(fcw, fcw16);
        size_t n0 = (size_t)G4H * XK0;
        k_wcat<<<(unsigned)((n0 + 255) / 256), 256, 0, stream>>>(Wi0, Wh0, wcat0, 1024);
        size_t n1 = (size_t)G4H * XK1;
        k_wcat<<<(unsigned)((n1 + 255) / 256), 256, 0, stream>>>(Wi1, Wh1, wcat1, 512);
        k_bias<<<(G4H + 255) / 256, 256, 0, stream>>>(bi0, bh0, bi1, bh1, bias0, bias1);
        k_init<<<(BB * HH + 255) / 256, 256, 0, stream>>>(init_h, init_c, h0, c0, h1, c1);
    }

    // enc_proj = encoder_outputs @ W1^T + b1   (M=4096, N=512, K=512), then (b,k,e)
    wmma_gemm<4, HH, false><<<dim3(1, (BB * TEE) / 64), 256, 0, stream>>>(
        enc16, w116, W1b, encproj, HH, HH);
    k_encT<<<(BB * HH * TEE + 255) / 256, 256, 0, stream>>>(encproj, encT);

    // ---- sequential decode: 4 kernels per step ----
    for (int t = 0; t < TDD; ++t) {
        attn_fused_kernel<<<BB, 256, 0, stream>>>(
            h1, c1, g1, hs16, h0, w2t, W2b, encT, encout, smask, vw, vb, ids, emb,
            xcat0, t);
        // gates layer 0: (16, 2048) = xcat0 (16,1536) @ wcat0^T
        wmma_gemm<1, XK0, false><<<dim3(G4H / 512, 1), 256, 0, stream>>>(
            xcat0, wcat0, bias0, g0, G4H, G4H);
        cell_kernel<<<(BB * HH) / 256, 256, 0, stream>>>(g0, h0, c0, h1, xcat1, 0, t);
        // gates layer 1: (16, 2048) = xcat1 (16,1024) @ wcat1^T
        wmma_gemm<1, XK1, false><<<dim3(G4H / 512, 1), 256, 0, stream>>>(
            xcat1, wcat1, bias1, g1, G4H, G4H);
    }
    // tail: layer-1 cell for t = T-1
    cell_kernel<<<(BB * HH) / 256, 256, 0, stream>>>(g1, h1, c1, nullptr, hs16, 1, TDD - 1);

    // ---- batched vocab projection: (1024, 50257) = hs (1024,512) @ fc_w^T ----
    // hs rows are ordered b*T+t, so row-major output == d_out (B,T,V).
    // NT stores: logits are write-once, keep fc_w_f16 resident in L2.
    wmma_gemm<4, HH, true><<<dim3(NPADFC / 512, (BB * TDD) / 64), 256, 0, stream>>>(
        hs16, fcw16, fcb, out, VVOC, VVOC);
}
